// SelfAttention_4767413698818
// MI455X (gfx1250) — compile-verified, matrix operand order checked by reading, not run on hardware
//
#include <hip/hip_runtime.h>
#include <hip/hip_bf16.h>

// ---------------------------------------------------------------------------
// CDNA5 (gfx1250, wave32) bf16-WMMA implementation of the reference
// "per-position head-mixing attention" block.
//
// qkv[b*4096+i, e] = sum_d x[b*4096+i, d] * w_qkv[e, d]       (bf16 WMMA)
// For position (b,n):  q[h,d] = qkv[b*4096 + h*256 + n/16, (n%16)*64 + d]
//                      k,v same with +1024 / +2048 column offsets.
// S = QK^T/8 (16x16, K=64), softmax rows, attn = P V (16x64, K=16).
// attn scattered back with the inverse mapping, then FC GEMM + bias (f32 out).
//
// GEMM blocking: 256-thread blocks (8 waves). Each wave owns a 32x64 tile
// (2 A-frags x 4 B-frags -> 8 WMMAs per K-step of 32; 12 b128 loads per step,
// i.e. 1.5 loads/WMMA).
// ---------------------------------------------------------------------------

typedef __bf16 bf16_t;
typedef __attribute__((ext_vector_type(16))) __bf16 bf16x16;
typedef __attribute__((ext_vector_type(8)))  __bf16 bf16x8;
typedef __attribute__((ext_vector_type(8)))  float  f32x8;

union Frag {
    bf16x16 v;
    bf16x8  h[2];
};

#define WMMA_BF16(A, B, C) \
    __builtin_amdgcn_wmma_f32_16x16x32_bf16(false, (A), false, (B), (short)0, (C), false, false)

// Fragment: lanes 0-15: row=lane, K = +0..7 (h[0]), +16..23 (h[1]);
//           lanes 16-31: row=lane-16, K = +8..15, +24..31.
__device__ __forceinline__ Frag ldfrag(const bf16_t* __restrict__ p) {
    Frag f;
    f.h[0] = *(const bf16x8*)(p);
    f.h[1] = *(const bf16x8*)(p + 16);
    return f;
}

// ---------------------------------------------------------------------------
// f32 -> bf16 conversion (grid-stride)
// ---------------------------------------------------------------------------
__global__ __launch_bounds__(256) void cvt_f32_bf16_kernel(const float* __restrict__ in,
                                                           bf16_t* __restrict__ out,
                                                           size_t n) {
    size_t i = (size_t)blockIdx.x * blockDim.x + threadIdx.x;
    size_t stride = (size_t)gridDim.x * blockDim.x;
    for (; i < n; i += stride) out[i] = (bf16_t)in[i];
}

// ---------------------------------------------------------------------------
// GEMM: C[m,n] = sum_k A[m,k] * W[n,k]   (A: MxK bf16, W: NxK bf16 row-major)
// Block tile: 256 rows x 64 cols. Wave w -> rows [blockIdx.x*256 + 32w, +32).
// ---------------------------------------------------------------------------
__global__ __launch_bounds__(256) void gemm_bf16out_kernel(const bf16_t* __restrict__ A,
                                                           const bf16_t* __restrict__ W,
                                                           bf16_t* __restrict__ C,
                                                           int M, int N, int K) {
    const int lane = threadIdx.x & 31;
    const int wave = threadIdx.x >> 5;
    const int row0 = blockIdx.x * 256 + wave * 32;
    const int col0 = blockIdx.y * 64;
    const int cl = lane & 15, lg = lane >> 4;

    const bf16_t* ap0 = A + (size_t)(row0 + cl) * K + (lg << 3);
    const bf16_t* ap1 = ap0 + (size_t)16 * K;
    const bf16_t* bp  = W + (size_t)(col0 + cl) * K + (lg << 3);

    f32x8 acc[2][4] = {};
#pragma unroll 2
    for (int ks = 0; ks < K; ks += 32) {
        __builtin_prefetch(ap0 + ks + 256, 0, 0);        // global_prefetch_b8
        __builtin_prefetch(ap1 + ks + 256, 0, 0);
        Frag a0 = ldfrag(ap0 + ks);
        Frag a1 = ldfrag(ap1 + ks);
        Frag b0 = ldfrag(bp + ks);
        Frag b1 = ldfrag(bp + (size_t)16 * K + ks);
        Frag b2 = ldfrag(bp + (size_t)32 * K + ks);
        Frag b3 = ldfrag(bp + (size_t)48 * K + ks);
        acc[0][0] = WMMA_BF16(a0.v, b0.v, acc[0][0]);
        acc[0][1] = WMMA_BF16(a0.v, b1.v, acc[0][1]);
        acc[0][2] = WMMA_BF16(a0.v, b2.v, acc[0][2]);
        acc[0][3] = WMMA_BF16(a0.v, b3.v, acc[0][3]);
        acc[1][0] = WMMA_BF16(a1.v, b0.v, acc[1][0]);
        acc[1][1] = WMMA_BF16(a1.v, b1.v, acc[1][1]);
        acc[1][2] = WMMA_BF16(a1.v, b2.v, acc[1][2]);
        acc[1][3] = WMMA_BF16(a1.v, b3.v, acc[1][3]);
    }

    // C/D layout: VGPR r -> row r (lanes 0-15) / r+8 (lanes 16-31), col = lane&15
#pragma unroll
    for (int mi = 0; mi < 2; ++mi)
#pragma unroll
        for (int r = 0; r < 8; ++r) {
            bf16_t* cp = C + (size_t)(row0 + 16 * mi + r + 8 * lg) * N + col0 + cl;
            cp[0]  = (bf16_t)acc[mi][0][r];
            cp[16] = (bf16_t)acc[mi][1][r];
            cp[32] = (bf16_t)acc[mi][2][r];
            cp[48] = (bf16_t)acc[mi][3][r];
        }
}

__global__ __launch_bounds__(256) void gemm_f32bias_kernel(const bf16_t* __restrict__ A,
                                                           const bf16_t* __restrict__ W,
                                                           const float* __restrict__ bias,
                                                           float* __restrict__ C,
                                                           int M, int N, int K) {
    const int lane = threadIdx.x & 31;
    const int wave = threadIdx.x >> 5;
    const int row0 = blockIdx.x * 256 + wave * 32;
    const int col0 = blockIdx.y * 64;
    const int cl = lane & 15, lg = lane >> 4;

    const bf16_t* ap0 = A + (size_t)(row0 + cl) * K + (lg << 3);
    const bf16_t* ap1 = ap0 + (size_t)16 * K;
    const bf16_t* bp  = W + (size_t)(col0 + cl) * K + (lg << 3);

    f32x8 acc[2][4] = {};
#pragma unroll 2
    for (int ks = 0; ks < K; ks += 32) {
        __builtin_prefetch(ap0 + ks + 256, 0, 0);
        __builtin_prefetch(ap1 + ks + 256, 0, 0);
        Frag a0 = ldfrag(ap0 + ks);
        Frag a1 = ldfrag(ap1 + ks);
        Frag b0 = ldfrag(bp + ks);
        Frag b1 = ldfrag(bp + (size_t)16 * K + ks);
        Frag b2 = ldfrag(bp + (size_t)32 * K + ks);
        Frag b3 = ldfrag(bp + (size_t)48 * K + ks);
        acc[0][0] = WMMA_BF16(a0.v, b0.v, acc[0][0]);
        acc[0][1] = WMMA_BF16(a0.v, b1.v, acc[0][1]);
        acc[0][2] = WMMA_BF16(a0.v, b2.v, acc[0][2]);
        acc[0][3] = WMMA_BF16(a0.v, b3.v, acc[0][3]);
        acc[1][0] = WMMA_BF16(a1.v, b0.v, acc[1][0]);
        acc[1][1] = WMMA_BF16(a1.v, b1.v, acc[1][1]);
        acc[1][2] = WMMA_BF16(a1.v, b2.v, acc[1][2]);
        acc[1][3] = WMMA_BF16(a1.v, b3.v, acc[1][3]);
    }

    const float bi0 = bias[col0 + cl];
    const float bi1 = bias[col0 + 16 + cl];
    const float bi2 = bias[col0 + 32 + cl];
    const float bi3 = bias[col0 + 48 + cl];
#pragma unroll
    for (int mi = 0; mi < 2; ++mi)
#pragma unroll
        for (int r = 0; r < 8; ++r) {
            float* cp = C + (size_t)(row0 + 16 * mi + r + 8 * lg) * N + col0 + cl;
            cp[0]  = acc[mi][0][r] + bi0;
            cp[16] = acc[mi][1][r] + bi1;
            cp[32] = acc[mi][2][r] + bi2;
            cp[48] = acc[mi][3][r] + bi3;
        }
}

// ---------------------------------------------------------------------------
// Attention: one wave per (b, n) position. 16x16 head-mixing attention.
// ---------------------------------------------------------------------------
__global__ __launch_bounds__(128) void attn_kernel(const bf16_t* __restrict__ qkvb,
                                                   bf16_t* __restrict__ attnb) {
    __shared__ __align__(16) float  plds[4][16][17];     // P: C-layout -> A-layout bounce
    __shared__ __align__(16) bf16_t vlds[4][16][72];     // V tile (16 heads x 64 dims), padded

    const int lane = threadIdx.x & 31;
    const int w    = threadIdx.x >> 5;
    const int p    = blockIdx.x * 4 + w;                 // 0..16383
    const int b    = p >> 12;
    const int n2   = p & 4095;
    const size_t ibase = (size_t)b * 4096 + (size_t)(n2 >> 4);
    const int jbase = (n2 & 15) << 6;
    const int cl = lane & 15;
    const int lg = lane >> 4;

    // Stage V tile into LDS: V[g][d] = qkvb[(ibase + g*256)*3072 + 2048 + jbase + d]
#pragma unroll
    for (int t = 0; t < 4; ++t) {
        int cid = t * 32 + lane;                         // 0..127 16B chunks
        int g   = cid >> 3;
        int o8  = (cid & 7) << 3;
        const bf16_t* src = qkvb + (ibase + (size_t)g * 256) * 3072 + 2048 + jbase + o8;
        *(bf16x8*)&vlds[w][g][o8] = *(const bf16x8*)src;
    }

    // S = Q K^T : A-frag rows = heads (q), B-frag cols = heads (k), K-dim = 64
    f32x8 s = {};
    const bf16_t* qp = qkvb + (ibase + (size_t)cl * 256) * 3072 + jbase + (lg << 3);
    const bf16_t* kp = qp + 1024;
#pragma unroll
    for (int ks = 0; ks < 64; ks += 32) {
        Frag a  = ldfrag(qp + ks);
        Frag bk = ldfrag(kp + ks);
        s = WMMA_BF16(a.v, bk.v, s);
    }

    // Softmax over g (columns). Row h = r + 8*lg lives across the 16 lanes of
    // this lane's half; masks 1/2/4/8 never cross the half boundary.
    float pr[8];
#pragma unroll
    for (int r = 0; r < 8; ++r) {
        float x = s[r] * 0.125f;                         // 1/sqrt(64)
        float m = x;
        m = fmaxf(m, __shfl_xor(m, 1));
        m = fmaxf(m, __shfl_xor(m, 2));
        m = fmaxf(m, __shfl_xor(m, 4));
        m = fmaxf(m, __shfl_xor(m, 8));
        float e = __expf(x - m);
        float su = e;
        su += __shfl_xor(su, 1);
        su += __shfl_xor(su, 2);
        su += __shfl_xor(su, 4);
        su += __shfl_xor(su, 8);
        pr[r] = e / su;
    }

    // Bounce P through LDS to convert C-layout -> A-layout
#pragma unroll
    for (int r = 0; r < 8; ++r) plds[w][r + 8 * lg][cl] = pr[r];
    __syncthreads();

    Frag pa = {};                                        // K = 16 padded to 32 with zeros
    {
        const int g0 = lg << 3;
        bf16x8 t;
#pragma unroll
        for (int i = 0; i < 8; ++i) t[i] = (bf16_t)plds[w][cl][g0 + i];
        pa.h[0] = t;
    }

    // attn = P @ V : 4 column chunks of 16
    f32x8 av[4];
#pragma unroll
    for (int dj = 0; dj < 4; ++dj) {
        Frag vb = {};
        const int g0 = lg << 3;
        const int d  = dj * 16 + cl;
        bf16x8 t;
#pragma unroll
        for (int i = 0; i < 8; ++i) t[i] = vlds[w][g0 + i][d];
        vb.h[0] = t;
        f32x8 z = {};
        av[dj] = WMMA_BF16(pa.v, vb.v, z);
    }

    // Inverse prep mapping: attnb[(ibase + h*256)*1024 + jbase + d]
#pragma unroll
    for (int dj = 0; dj < 4; ++dj)
#pragma unroll
        for (int r = 0; r < 8; ++r) {
            const int h = r + 8 * lg;
            attnb[(ibase + (size_t)h * 256) * 1024 + jbase + dj * 16 + cl] =
                (bf16_t)av[dj][r];
        }
}

// ---------------------------------------------------------------------------
// Launch
// ---------------------------------------------------------------------------
extern "C" void kernel_launch(void* const* d_in, const int* in_sizes, int n_in,
                              void* d_out, int out_size, void* d_ws, size_t ws_size,
                              hipStream_t stream) {
    constexpr int Bn = 4, Nn = 4096, Dn = 1024;
    constexpr int Mrows = Bn * Nn;                       // 16384
    const float* x     = (const float*)d_in[0];          // (4,4096,1024)
    const float* w_qkv = (const float*)d_in[1];          // (3072,1024)
    const float* w_fc  = (const float*)d_in[2];          // (1024,1024)
    const float* b_fc  = (const float*)d_in[3];          // (1024,)
    float* out = (float*)d_out;

    // Workspace carve-up (bf16): x 32MB | w_qkv 6MB | w_fc 2MB | qkv 96MB | attn 32MB
    char* ws = (char*)d_ws;
    bf16_t* xb    = (bf16_t*)ws;                 ws += (size_t)Mrows * Dn * 2;
    bf16_t* wqkvb = (bf16_t*)ws;                 ws += (size_t)3 * Dn * Dn * 2;
    bf16_t* wfcb  = (bf16_t*)ws;                 ws += (size_t)Dn * Dn * 2;
    bf16_t* qkvb  = (bf16_t*)ws;                 ws += (size_t)Mrows * 3 * Dn * 2;
    bf16_t* attnb = (bf16_t*)ws;

    // 0) f32 -> bf16
    cvt_f32_bf16_kernel<<<4096, 256, 0, stream>>>(x, xb, (size_t)Mrows * Dn);
    cvt_f32_bf16_kernel<<<1536, 256, 0, stream>>>(w_qkv, wqkvb, (size_t)3 * Dn * Dn);
    cvt_f32_bf16_kernel<<<512, 256, 0, stream>>>(w_fc, wfcb, (size_t)Dn * Dn);

    // 1) qkv = x @ w_qkv^T    (16384 x 3072, K=1024)
    gemm_bf16out_kernel<<<dim3(Mrows / 256, 3 * Dn / 64), 256, 0, stream>>>(
        xb, wqkvb, qkvb, Mrows, 3 * Dn, Dn);

    // 2) per-position head-mixing attention (one wave per position)
    attn_kernel<<<Mrows / 4, 128, 0, stream>>>(qkvb, attnb);

    // 3) out = attn @ w_fc^T + b_fc   (16384 x 1024, K=1024, f32 out)
    gemm_f32bias_kernel<<<dim3(Mrows / 256, Dn / 64), 256, 0, stream>>>(
        attnb, wfcb, b_fc, out, Mrows, Dn, Dn);
}
